// SetUpConv_78426102825590
// MI455X (gfx1250) — matrix operand order repaired and checked by reading, not compiled, not run on hardware
//
#include <hip/hip_runtime.h>

// ---------------------------------------------------------------------------
// SetUpConv for MI455X (gfx1250, wave32, WMMA).
// Pipeline: KNN -> gather/pack(f16, WMMA-tiled) -> GEMM(wmma f16->f32) ->
//           GroupNorm stats -> normalize+LReLU -> GEMM -> GN -> maxK+concat ->
//           GEMM -> GN -> f32 output.
// ---------------------------------------------------------------------------

typedef __attribute__((ext_vector_type(16))) _Float16 v16h;
typedef __attribute__((ext_vector_type(8)))  float    v8f;

constexpr int BATCH = 4;
constexpr int N1    = 1024;
constexpr int N2    = 4096;
constexpr int KNN   = 16;
constexpr int C1    = 256;
constexpr int C2    = 128;
constexpr int P1    = BATCH * N2 * KNN;   // 262144 points for MLP1
constexpr int P2    = BATCH * N2;         // 16384 points for MLP2
constexpr int NS1   = 9;                  // ci=259 padded to 288 = 9*32
constexpr int NS2   = 8;                  // ci=256
constexpr int NS3   = 12;                 // ci=384
constexpr int NSO   = 8;                  // co=256 (output packed steps)
constexpr int COT   = 16;                 // 256/16 co-tiles

// ---- packed-layout helpers (WMMA fragments, contiguous k = 16h + s order) -------------------
// B-matrix (32x16 f16): VGPR elems 0..15; lanes 0-15 hold K0..15 (N=lane),
// lanes 16-31 hold K16..31.  Row = (tile*nSteps + step)*32 + lane, 16 halfs.
__device__ __forceinline__ size_t packB(int p, int c, int nSteps) {
  int tile = p >> 4, np = p & 15;
  int step = c >> 5, r = c & 31;
  int lane = np + (r & 16);
  int elem = r & 15;
  return ((size_t)(tile * nSteps + step) * 32 + lane) * 16 + elem;
}
// A-matrix (16x32 f16): lanes 0-15: M=lane, elems 0..15=K0..15;
// lanes 16-31: M=lane-16, elems 0..15=K16..31.
__device__ __forceinline__ size_t packA(int co, int ci, int nSteps) {
  int coT = co >> 4, m = co & 15;
  int step = ci >> 5, r = ci & 31;
  int hi   = (r >> 4) & 1;
  int lane = m + (hi << 4);
  int elem = r & 15;
  return ((size_t)(coT * nSteps + step) * 32 + lane) * 16 + elem;
}

// ---------------------------------------------------------------------------
__global__ void zero_kernel(float* p, int n) {
  int i = blockIdx.x * 256 + threadIdx.x;
  if (i < n) p[i] = 0.0f;
}

// ---- KNN: one thread per query, targets staged in LDS ----------------------
__global__ void knn_kernel(const float* __restrict__ xyz1,
                           const float* __restrict__ xyz2,
                           int* __restrict__ idx) {
  __shared__ float sx[N1], sy[N1], sz[N1];
  int b  = blockIdx.x >> 4;
  int qb = blockIdx.x & 15;
  const float* t = xyz1 + (size_t)b * 3 * N1;
  for (int i = threadIdx.x; i < N1; i += 256) {
    sx[i] = t[i]; sy[i] = t[N1 + i]; sz[i] = t[2 * N1 + i];
  }
  __syncthreads();
  int n2 = qb * 256 + threadIdx.x;
  float qx = xyz2[((size_t)b * 3 + 0) * N2 + n2];
  float qy = xyz2[((size_t)b * 3 + 1) * N2 + n2];
  float qz = xyz2[((size_t)b * 3 + 2) * N2 + n2];
  float bd[KNN]; int bi[KNN];
#pragma unroll
  for (int k = 0; k < KNN; ++k) { bd[k] = 3.4e38f; bi[k] = 0; }
  for (int j = 0; j < N1; ++j) {
    float dx = sx[j] - qx, dy = sy[j] - qy, dz = sz[j] - qz;
    float d = dx * dx + dy * dy + dz * dz;
    if (d < bd[KNN - 1]) {
      float cd = d; int ci = j;
#pragma unroll
      for (int k = 0; k < KNN; ++k) {
        if (cd < bd[k]) { float td = bd[k]; int ti = bi[k];
                          bd[k] = cd; bi[k] = ci; cd = td; ci = ti; }
      }
    }
  }
  int* op = idx + ((size_t)(b * N2 + n2)) * KNN;
#pragma unroll
  for (int k = 0; k < KNN; ++k) op[k] = bi[k];
}

// ---- pack weights (f32 row-major -> A-fragment f16, zero-padded) -----------
__global__ void pack_w_kernel(const float* __restrict__ W, _Float16* __restrict__ Wp,
                              int ciActual, int ciPad, int nSteps) {
  int t = blockIdx.x * 256 + threadIdx.x;
  if (t >= 256 * ciPad) return;
  int co = t / ciPad, ci = t % ciPad;
  float v = (ci < ciActual) ? W[(size_t)co * ciActual + ci] : 0.0f;
  Wp[packA(co, ci, nSteps)] = (_Float16)v;
}

// ---- build MLP1 input: gather feat1 + relative xyz, pack to B layout -------
__global__ void build_x1_kernel(const float* __restrict__ feat1,
                                const float* __restrict__ xyz1,
                                const float* __restrict__ xyz2,
                                const int* __restrict__ idx,
                                _Float16* __restrict__ Xp) {
  int p = blockIdx.x * 256 + threadIdx.x;   // < P1
  int b   = p >> 16;                        // / (N2*KNN)
  int rem = p & 65535;
  int n2  = rem >> 4;
  int j   = idx[p];
  const float* f1 = feat1 + (size_t)b * C1 * N1;
  for (int c = 0; c < C1; ++c)
    Xp[packB(p, c, NS1)] = (_Float16)f1[(size_t)c * N1 + j];
#pragma unroll
  for (int d = 0; d < 3; ++d) {
    float v = xyz1[((size_t)b * 3 + d) * N1 + j] - xyz2[((size_t)b * 3 + d) * N2 + n2];
    Xp[packB(p, C1 + d, NS1)] = (_Float16)v;
  }
  for (int c = C1 + 3; c < NS1 * 32; ++c)
    Xp[packB(p, c, NS1)] = (_Float16)0.0f;
}

// ---- GEMM: one wave -> 16co x 32pt (two 16x16 tiles sharing A) -------------
template <int NS>
__global__ void gemm_kernel(const _Float16* __restrict__ Xp,
                            const _Float16* __restrict__ Wp,
                            const float* __restrict__ bias,
                            _Float16* __restrict__ Yp,
                            int coTiles, int pTiles, int nStepsOut) {
  int tid  = blockIdx.x * 256 + threadIdx.x;
  int wave = tid >> 5;
  int lane = tid & 31;
  int totWaves = coTiles * (pTiles >> 1);
  if (wave >= totWaves) return;
  int coT = wave % coTiles;
  int pT0 = (wave / coTiles) * 2;

  const v16h* Wv = (const v16h*)Wp;
  const v16h* Xv = (const v16h*)Xp;
  size_t aBase  = (size_t)coT * NS * 32 + lane;
  size_t b0Base = (size_t)pT0 * NS * 32 + lane;
  size_t b1Base = b0Base + (size_t)NS * 32;

  v8f acc0 = {}; v8f acc1 = {};
#pragma unroll
  for (int s = 0; s < NS; ++s) {
    v16h a  = Wv[aBase  + (size_t)s * 32];
    v16h b0 = Xv[b0Base + (size_t)s * 32];
    v16h b1 = Xv[b1Base + (size_t)s * 32];
    acc0 = __builtin_amdgcn_wmma_f32_16x16x32_f16(false, a, false, b0,
                                                  (short)0, acc0, false, false);
    acc1 = __builtin_amdgcn_wmma_f32_16x16x32_f16(false, a, false, b1,
                                                  (short)0, acc1, false, false);
  }
  int np = lane & 15, hi = lane >> 4;
  int p0 = pT0 * 16 + np, p1 = p0 + 16;
#pragma unroll
  for (int r = 0; r < 8; ++r) {
    int co = coT * 16 + r + hi * 8;          // C/D layout: lanes16-31 -> M=r+8
    float bv = bias[co];
    Yp[packB(p0, co, nStepsOut)] = (_Float16)(acc0[r] + bv);
    Yp[packB(p1, co, nStepsOut)] = (_Float16)(acc1[r] + bv);
  }
}

// ---- GroupNorm stats: sum / sumsq per (batch, group-of-16-channels) --------
__global__ void stats_kernel(const _Float16* __restrict__ Y, float* __restrict__ stats,
                             int nSteps, int pointsPerB, int chunks) {
  __shared__ float s1[256], s2[256];
  int ngroups = nSteps * 2;
  int bg = blockIdx.x / chunks;
  int ck = blockIdx.x % chunks;
  int b = bg / ngroups, g = bg % ngroups;
  int tilesPerB = pointsPerB >> 4;
  int tilesPerChunk = tilesPerB / chunks;
  int tile0 = b * tilesPerB + ck * tilesPerChunk;
  int rows = tilesPerChunk * 16;
  float sum = 0.0f, sq = 0.0f;
  for (int i = threadIdx.x; i < rows; i += 256) {
    int t = tile0 + (i >> 4);
    int laneRow = i & 15;
    const _Float16* ptr =
        Y + ((size_t)(t * nSteps + (g >> 1)) * 32 + ((g & 1) << 4) + laneRow) * 16;
#pragma unroll
    for (int e = 0; e < 16; ++e) { float v = (float)ptr[e]; sum += v; sq += v * v; }
  }
  s1[threadIdx.x] = sum; s2[threadIdx.x] = sq;
  __syncthreads();
  for (int o = 128; o > 0; o >>= 1) {
    if (threadIdx.x < o) { s1[threadIdx.x] += s1[threadIdx.x + o];
                           s2[threadIdx.x] += s2[threadIdx.x + o]; }
    __syncthreads();
  }
  if (threadIdx.x == 0) {
    atomicAdd(&stats[2 * (b * ngroups + g)],     s1[0]);
    atomicAdd(&stats[2 * (b * ngroups + g) + 1], s2[0]);
  }
}

// ---- normalize + LeakyReLU in packed layout (in place) ---------------------
__global__ void gn_kernel(_Float16* __restrict__ Y, const float* __restrict__ stats,
                          const float* __restrict__ gamma, const float* __restrict__ beta,
                          int nSteps, int pointsPerB, int totalRows) {
  int row = blockIdx.x * 256 + threadIdx.x;
  if (row >= totalRows) return;
  int rowsPerTile = nSteps * 32;
  int tile = row / rowsPerTile;
  int rem  = row % rowsPerTile;
  int step = rem >> 5, lane = rem & 31;
  int cbase = step * 32 + (lane & 16);
  int g = cbase >> 4;
  int ngroups = nSteps * 2;
  int b = (tile * 16) / pointsPerB;
  float s = stats[2 * (b * ngroups + g)];
  float q = stats[2 * (b * ngroups + g) + 1];
  float cnt = 16.0f * (float)pointsPerB;
  float mu = s / cnt;
  float var = q / cnt - mu * mu;
  float rs = rsqrtf(var + 1e-5f);
  _Float16* ptr = Y + (size_t)row * 16;
#pragma unroll
  for (int e = 0; e < 16; ++e) {
    int c = cbase + e;
    float v = (float)ptr[e];
    v = (v - mu) * rs * gamma[c] + beta[c];
    v = (v >= 0.0f) ? v : 0.1f * v;
    ptr[e] = (_Float16)v;
  }
}

// ---- max over K (=16 points in one tile) + concat feat2 -> MLP2 input ------
__global__ void maxconcat_kernel(const _Float16* __restrict__ Y2,
                                 const float* __restrict__ feat2,
                                 _Float16* __restrict__ X2p) {
  int t = blockIdx.x * 256 + threadIdx.x;    // < P2 * 384
  int pn = t / 384;                          // b*N2+n2
  int c2 = t % 384;
  int b = pn >> 12, n2 = pn & 4095;
  float v;
  if (c2 < C2) {
    v = feat2[((size_t)b * C2 + c2) * N2 + n2];
  } else {
    int c = c2 - C2;
    int step = c >> 5, r = c & 31;
    size_t base = ((size_t)(pn * NSO + step) * 32 + (r & 16)) * 16 + (r & 15);
    v = -3.4e38f;
#pragma unroll
    for (int k = 0; k < KNN; ++k) v = fmaxf(v, (float)Y2[base + (size_t)k * 16]);
  }
  X2p[packB(pn, c2, NS3)] = (_Float16)v;
}

// ---- final normalize + LeakyReLU, write f32 (B,256,N2) ---------------------
__global__ void gn_final_kernel(const _Float16* __restrict__ Y,
                                const float* __restrict__ stats,
                                const float* __restrict__ gamma,
                                const float* __restrict__ beta,
                                float* __restrict__ out, int totalRows) {
  int row = blockIdx.x * 256 + threadIdx.x;
  if (row >= totalRows) return;
  int tile = row >> 8;                       // 8 steps * 32 lanes
  int rem  = row & 255;
  int step = rem >> 5, lane = rem & 31;
  int cbase = step * 32 + (lane & 16);
  int g = cbase >> 4;
  int np = lane & 15;
  int p = tile * 16 + np;
  int b = p >> 12, n2 = p & 4095;
  float s = stats[2 * (b * 16 + g)];
  float q = stats[2 * (b * 16 + g) + 1];
  float cnt = 16.0f * (float)N2;
  float mu = s / cnt;
  float var = q / cnt - mu * mu;
  float rs = rsqrtf(var + 1e-5f);
  const _Float16* ptr = Y + (size_t)row * 16;
#pragma unroll
  for (int e = 0; e < 16; ++e) {
    int c = cbase + e;
    float v = (float)ptr[e];
    v = (v - mu) * rs * gamma[c] + beta[c];
    v = (v >= 0.0f) ? v : 0.1f * v;
    out[((size_t)(b * 256 + c)) * N2 + n2] = v;
  }
}

// ---------------------------------------------------------------------------
extern "C" void kernel_launch(void* const* d_in, const int* in_sizes, int n_in,
                              void* d_out, int out_size, void* d_ws, size_t ws_size,
                              hipStream_t stream) {
  const float* xyz1  = (const float*)d_in[0];
  const float* xyz2  = (const float*)d_in[1];
  const float* feat1 = (const float*)d_in[2];
  const float* feat2 = (const float*)d_in[3];
  const float* W1a   = (const float*)d_in[4];   // (256,259)
  const float* W1b   = (const float*)d_in[5];   // (256,256)
  const float* b1a   = (const float*)d_in[6];
  const float* b1b   = (const float*)d_in[7];
  const float* g1a   = (const float*)d_in[8];
  const float* g1b   = (const float*)d_in[9];
  const float* be1a  = (const float*)d_in[10];
  const float* be1b  = (const float*)d_in[11];
  const float* W2a   = (const float*)d_in[12];  // (256,384)
  const float* b2a   = (const float*)d_in[13];
  const float* g2a   = (const float*)d_in[14];
  const float* be2a  = (const float*)d_in[15];
  float* out = (float*)d_out;

  char* w = (char*)d_ws;
  auto carve = [&](size_t bytes) -> void* {
    void* p = (void*)w;
    w += (bytes + 255) & ~(size_t)255;
    return p;
  };
  int*      idx   = (int*)     carve((size_t)P1 * 4);
  _Float16* W1p   = (_Float16*)carve((size_t)COT * NS1 * 512 * 2);
  _Float16* W2p   = (_Float16*)carve((size_t)COT * NS2 * 512 * 2);
  _Float16* W3p   = (_Float16*)carve((size_t)COT * NS3 * 512 * 2);
  float*    stats = (float*)   carve(3 * 64 * 2 * 4);   // 3 layers x (4b x 16g) x {sum,sq}
  _Float16* X1p   = (_Float16*)carve((size_t)P1 * NS1 * 32 * 2);   // 151 MB
  _Float16* Y1    = (_Float16*)carve((size_t)P1 * 256 * 2);        // 134 MB
  _Float16* Y2    = X1p;   // alias: X1p dead after GEMM1 (134 MB <= 151 MB)
  _Float16* X2p   = (_Float16*)carve((size_t)P2 * NS3 * 32 * 2);   // 12.6 MB
  _Float16* Y3    = (_Float16*)carve((size_t)P2 * 256 * 2);        // 8.4 MB
  float* stats1 = stats, * stats2 = stats + 128, * stats3 = stats + 256;

  zero_kernel<<<2, 256, 0, stream>>>(stats, 384);
  knn_kernel<<<BATCH * (N2 / 256), 256, 0, stream>>>(xyz1, xyz2, idx);

  pack_w_kernel<<<(256 * 288) / 256, 256, 0, stream>>>(W1a, W1p, 259, 288, NS1);
  pack_w_kernel<<<(256 * 256) / 256, 256, 0, stream>>>(W1b, W2p, 256, 256, NS2);
  pack_w_kernel<<<(256 * 384) / 256, 256, 0, stream>>>(W2a, W3p, 384, 384, NS3);

  build_x1_kernel<<<P1 / 256, 256, 0, stream>>>(feat1, xyz1, xyz2, idx, X1p);

  // MLP1 layer 1: 259->256 over 262144 points
  gemm_kernel<NS1><<<(COT * (P1 / 32)) * 32 / 256, 256, 0, stream>>>(
      X1p, W1p, b1a, Y1, COT, P1 / 16, NSO);
  stats_kernel<<<BATCH * 16 * 64, 256, 0, stream>>>(Y1, stats1, NSO, N2 * KNN, 64);
  gn_kernel<<<(P1 / 16) * NSO * 32 / 256, 256, 0, stream>>>(
      Y1, stats1, g1a, be1a, NSO, N2 * KNN, (P1 / 16) * NSO * 32);

  // MLP1 layer 2: 256->256
  gemm_kernel<NS2><<<(COT * (P1 / 32)) * 32 / 256, 256, 0, stream>>>(
      Y1, W2p, b1b, Y2, COT, P1 / 16, NSO);
  stats_kernel<<<BATCH * 16 * 64, 256, 0, stream>>>(Y2, stats2, NSO, N2 * KNN, 64);
  gn_kernel<<<(P1 / 16) * NSO * 32 / 256, 256, 0, stream>>>(
      Y2, stats2, g1b, be1b, NSO, N2 * KNN, (P1 / 16) * NSO * 32);

  // max over K + concat feat2 -> packed MLP2 input (384 ch)
  maxconcat_kernel<<<(P2 * 384) / 256, 256, 0, stream>>>(Y2, feat2, X2p);

  // MLP2: 384->256 over 16384 points
  gemm_kernel<NS3><<<(COT * (P2 / 32)) * 32 / 256, 256, 0, stream>>>(
      X2p, W3p, b2a, Y3, COT, P2 / 16, NSO);
  stats_kernel<<<BATCH * 16 * 16, 256, 0, stream>>>(Y3, stats3, NSO, N2, 16);
  gn_final_kernel<<<(P2 / 16) * NSO * 32 / 256, 256, 0, stream>>>(
      Y3, stats3, g2a, be2a, out, (P2 / 16) * NSO * 32);
}
